// CentroidHead_56324201120171
// MI455X (gfx1250) — matrix operand order repaired and checked by reading, not compiled, not run on hardware
//
#include <hip/hip_runtime.h>
#include <hip/hip_bf16.h>
#include <math.h>
#include <stdint.h>

typedef __attribute__((ext_vector_type(16))) _Float16 v16h;
typedef __attribute__((ext_vector_type(8)))  float    v8f;

#define TAU 0.1f
#define MIN_SCORE 0.5f
#define KTOP 256
#define GDIM 64
#define NB 4
#define LATENT 256
#define NCLS 20
#define LUTSZ (NB * GDIM * GDIM * GDIM)

// 7-point stencil (R=1 -> only offsets with |.|^2 <= 1.21)
__device__ __constant__ int OFX[7] = {-1, 0, 0, 0, 0, 0, 1};
__device__ __constant__ int OFY[7] = { 0,-1, 0, 0, 0, 1, 0};
__device__ __constant__ int OFZ[7] = { 0, 0,-1, 0, 1, 0, 0};

// ---------------- fills ----------------
__global__ void fill_i32(int* p, int v, int n) {
    for (int t = blockIdx.x * blockDim.x + threadIdx.x; t < n; t += gridDim.x * blockDim.x) p[t] = v;
}
__global__ void fill_f32(float* p, float v, int n) {
    for (int t = blockIdx.x * blockDim.x + threadIdx.x; t < n; t += gridDim.x * blockDim.x) p[t] = v;
}

// ---------------- GEMV via WMMA (split f16 hi/lo, 3 products ~ f32 accuracy) ----
// One wave32 per 16-row tile; 8 K-chunks of 32; D column 0 = row dot-products.
//
// A (16-bit 16x32, per ISA): lanes 0-15 = rows, elements 0..7 -> K=kbase+half*8+e,
//                            elements 8..15 -> K=kbase+16+half*8+(e-8)
// B (16-bit 32x16): same K order as A: elements 0..7 -> K=kbase+half*8+e,
//                            elements 8..15 -> K=kbase+16+half*8+(e-8).
// D: column 0 lives in lane 0 (M=0..7 over acc[0..7]) and lane 16 (M=8..15).
__global__ __launch_bounds__(256) void gemv_wmma(const float* __restrict__ feats,
                                                 const float* __restrict__ w,
                                                 float* __restrict__ dotv,
                                                 int n, int ntiles) {
    int wave = threadIdx.x >> 5;
    int tile = blockIdx.x * 8 + wave;
    if (tile >= ntiles) return;                 // wave-uniform exit: EXEC stays all-1s
    int lane = threadIdx.x & 31;
    int half = lane >> 4;                       // K-half select
    int m    = lane & 15;                       // row within tile
    int row  = tile * 16 + m;
    int rrow = row < n ? row : n - 1;           // tail-safe loads (stores guarded)

    const float* rbase = feats + (size_t)rrow * LATENT + half * 8;

    v8f acc = {};
    v16h b_hi, b_lo;
#pragma unroll
    for (int e = 0; e < 16; ++e) { b_hi[e] = (_Float16)0; b_lo[e] = (_Float16)0; }

    for (int s = 0; s < 8; ++s) {
        int kbase = s * 32;
        const float* arow = rbase + kbase;
        __builtin_prefetch(arow + 64, 0, 0);    // global_prefetch_b8: two chunks ahead

        // A: two 32B-aligned 8-float groups -> 4x global_load_b128
        const float4* a0 = reinterpret_cast<const float4*>(arow);
        const float4* a1 = reinterpret_cast<const float4*>(arow + 16);
        float4 p0 = a0[0], p1 = a0[1];
        float4 q0 = a1[0], q1 = a1[1];
        float af[16] = {p0.x, p0.y, p0.z, p0.w, p1.x, p1.y, p1.z, p1.w,
                        q0.x, q0.y, q0.z, q0.w, q1.x, q1.y, q1.z, q1.w};
        v16h a_hi, a_lo;
#pragma unroll
        for (int e = 0; e < 16; ++e) {
            _Float16 h = (_Float16)af[e];
            a_hi[e] = h;
            a_lo[e] = (_Float16)(af[e] - (float)h);
        }

        if (m == 0) {                            // only B column 0 carries the weights
            const float* wrow = w + kbase + half * 8;
#pragma unroll
            for (int e = 0; e < 16; ++e) {
                float wv = wrow[e < 8 ? e : e + 8];
                _Float16 h = (_Float16)wv;
                b_hi[e] = h;
                b_lo[e] = (_Float16)(wv - (float)h);
            }
        }
        acc = __builtin_amdgcn_wmma_f32_16x16x32_f16(false, a_hi, false, b_hi, (short)0, acc, false, false);
        acc = __builtin_amdgcn_wmma_f32_16x16x32_f16(false, a_lo, false, b_hi, (short)0, acc, false, false);
        acc = __builtin_amdgcn_wmma_f32_16x16x32_f16(false, a_hi, false, b_lo, (short)0, acc, false, false);
    }
    if (m == 0) {
        int base = tile * 16 + half * 8;
#pragma unroll
        for (int r = 0; r < 8; ++r)
            if (base + r < n) dotv[base + r] = acc[r];
    }
}

// ---------------- sigmoid * (1 - softmax0) ----------------
__global__ void head_kernel(const float* __restrict__ dotv, const float* __restrict__ sem,
                            const float* __restrict__ conv_b,
                            float* __restrict__ out_cs, float* __restrict__ scores, int n) {
    int i = blockIdx.x * blockDim.x + threadIdx.x;
    if (i >= n) return;
    float z = dotv[i] + conv_b[0];
    float s = 1.0f / (1.0f + expf(-z));
    const float* l = sem + (size_t)i * NCLS;
    float mx = l[0];
#pragma unroll
    for (int c = 1; c < NCLS; ++c) mx = fmaxf(mx, l[c]);
    float sum = 0.0f, e0 = 0.0f;
#pragma unroll
    for (int c = 0; c < NCLS; ++c) {
        float e = expf(l[c] - mx);
        sum += e;
        if (c == 0) e0 = e;
    }
    float cs = s * (1.0f - e0 / sum);
    out_cs[i] = cs;
    scores[i] = cs;
}

// ---------------- dense hash LUT ----------------
__global__ void scatter_lut(const int* __restrict__ coords, int* __restrict__ lut, int n) {
    int i = blockIdx.x * blockDim.x + threadIdx.x;
    if (i >= n) return;
    int b = coords[4 * i], x = coords[4 * i + 1], y = coords[4 * i + 2], z = coords[4 * i + 3];
    lut[((b * GDIM + x) * GDIM + y) * GDIM + z] = i;
}

// ---------------- peak detection + candidate compaction ----------------
__global__ void peak_kernel(const int* __restrict__ coords, const float* __restrict__ scores,
                            const int* __restrict__ lut, int* __restrict__ slot,
                            int* __restrict__ cand, int* __restrict__ counters, int n) {
    int i = blockIdx.x * blockDim.x + threadIdx.x;
    if (i >= n) return;
    slot[i] = -1;
    float si = scores[i];
    bool  mi = si > TAU;
    int b = coords[4 * i], cx = coords[4 * i + 1], cy = coords[4 * i + 2], cz = coords[4 * i + 3];
    float nmax = -INFINITY;
#pragma unroll
    for (int o = 0; o < 7; ++o) {
        int x = cx + OFX[o], y = cy + OFY[o], z = cz + OFZ[o];
        if (x < 0 || x >= GDIM || y < 0 || y >= GDIM || z < 0 || z >= GDIM) continue;
        int j = lut[((b * GDIM + x) * GDIM + y) * GDIM + z];
        if (j < 0) continue;
        float sj = scores[j];
        if (mi && sj > TAU) nmax = fmaxf(nmax, sj);
    }
    bool peak = mi && (si >= nmax - 1e-6f) && (si >= MIN_SCORE);
    if (peak) {
        int p = atomicAdd(&counters[0], 1);
        cand[p] = i;
    }
}

// ---------------- top-K: histogram -> threshold -> compact -> exact rank ----------------
__global__ void hist_kernel(const int* __restrict__ cand, const float* __restrict__ scores,
                            const int* __restrict__ counters, int* __restrict__ hist) {
    int M = counters[0];
    for (int t = blockIdx.x * blockDim.x + threadIdx.x; t < M; t += gridDim.x * blockDim.x) {
        unsigned bits = __float_as_uint(scores[cand[t]]);   // positive floats: monotonic bits
        atomicAdd(&hist[bits >> 20], 1);
    }
}

__global__ void thresh_kernel(const int* __restrict__ hist, int* __restrict__ counters) {
    int cum = 0, T = 0;
    for (int bkt = 4095; bkt >= 0; --bkt) {
        cum += hist[bkt];
        T = bkt;
        if (cum >= KTOP) break;
    }
    counters[1] = T;
}

__global__ void compact_kernel(const int* __restrict__ cand, const float* __restrict__ scores,
                               int* __restrict__ counters, int* __restrict__ sel) {
    int M = counters[0];
    unsigned T = (unsigned)counters[1];
    for (int t = blockIdx.x * blockDim.x + threadIdx.x; t < M; t += gridDim.x * blockDim.x) {
        int i = cand[t];
        if ((__float_as_uint(scores[i]) >> 20) >= T) {
            int p = atomicAdd(&counters[2], 1);
            sel[p] = i;
        }
    }
}

// composite = (score_bits << 32) | (~idx): descending score, ascending index == jax top_k order
__global__ void rank_kernel(const int* __restrict__ sel, const float* __restrict__ scores,
                            const int* __restrict__ coords, const int* __restrict__ counters,
                            int* __restrict__ slot, float* __restrict__ out,
                            size_t o1, size_t o3, size_t o4) {
    int Ms = counters[2];
    for (int t = blockIdx.x * blockDim.x + threadIdx.x; t < Ms; t += gridDim.x * blockDim.x) {
        int i = sel[t];
        unsigned long long comp =
            ((unsigned long long)__float_as_uint(scores[i]) << 32) | (unsigned)(0xFFFFFFFFu - (unsigned)i);
        int rank = 0;
        for (int u = 0; u < Ms; ++u) {
            int j = sel[u];
            unsigned long long cj =
                ((unsigned long long)__float_as_uint(scores[j]) << 32) | (unsigned)(0xFFFFFFFFu - (unsigned)j);
            rank += (cj > comp) ? 1 : 0;
        }
        if (rank < KTOP) {
            slot[i] = rank;
            out[o3 + rank] = scores[i];
            out[o4 + rank] = 1.0f;
            out[o1 + (size_t)rank * 4 + 0] = (float)coords[4 * i + 0];
            out[o1 + (size_t)rank * 4 + 1] = (float)coords[4 * i + 1];
            out[o1 + (size_t)rank * 4 + 2] = (float)coords[4 * i + 2];
            out[o1 + (size_t)rank * 4 + 3] = (float)coords[4 * i + 3];
        }
    }
}

// ---------------- ownership + feature accumulation ----------------
__global__ void owner_kernel(const int* __restrict__ coords, const float* __restrict__ scores,
                             const int* __restrict__ lut, const int* __restrict__ slot,
                             const float* __restrict__ feats,
                             float* __restrict__ fsum, float* __restrict__ cnt, int n) {
    int i = blockIdx.x * blockDim.x + threadIdx.x;
    if (i >= n) return;
    float si = scores[i];
    if (!(si > TAU)) return;
    int b = coords[4 * i], cx = coords[4 * i + 1], cy = coords[4 * i + 2], cz = coords[4 * i + 3];
    int owner = -1;
#pragma unroll
    for (int o = 0; o < 7; ++o) {
        int x = cx + OFX[o], y = cy + OFY[o], z = cz + OFZ[o];
        if (x < 0 || x >= GDIM || y < 0 || y >= GDIM || z < 0 || z >= GDIM) continue;
        int j = lut[((b * GDIM + x) * GDIM + y) * GDIM + z];
        if (j < 0) continue;
        if (scores[j] > TAU) {
            int sl = slot[j];
            owner = sl > owner ? sl : owner;
        }
    }
    if (owner >= 0) {
        atomicAdd(&cnt[owner], 1.0f);
        const float* f = feats + (size_t)i * LATENT;
        for (int k = 0; k < LATENT; ++k) atomicAdd(&fsum[(size_t)owner * LATENT + k], f[k]);
    }
}

__global__ void mean_kernel(const float* __restrict__ fsum, const float* __restrict__ cnt,
                            float* __restrict__ out_mean) {
    int t = blockIdx.x * blockDim.x + threadIdx.x;
    if (t >= KTOP * LATENT) return;
    out_mean[t] = fsum[t] / fmaxf(cnt[t >> 8], 1.0f);
}

extern "C" void kernel_launch(void* const* d_in, const int* in_sizes, int n_in,
                              void* d_out, int out_size, void* d_ws, size_t ws_size,
                              hipStream_t stream) {
    const int*   coords = (const int*)d_in[0];
    const float* feats  = (const float*)d_in[1];
    const float* sem    = (const float*)d_in[2];
    const float* conv_w = (const float*)d_in[3];
    const float* conv_b = (const float*)d_in[4];
    float* out = (float*)d_out;

    const int n = in_sizes[0] / 4;
    const int ntiles = (n + 15) / 16;

    // output layout (flat f32, return order)
    size_t o1 = (size_t)n;                 // peak_coords  K*4
    size_t o2 = o1 + 4 * KTOP;             // feat_mean    K*256
    size_t o3 = o2 + (size_t)KTOP * LATENT;// confidence   K
    size_t o4 = o3 + KTOP;                 // peak_valid   K

    // workspace carve-up (256B aligned)
    char*  base = (char*)d_ws;
    size_t off = 0;
    auto take = [&](size_t bytes) -> char* {
        char* r = base + off;
        off += (bytes + 255) & ~(size_t)255;
        return r;
    };
    float* dotv     = (float*)take((size_t)n * 4);
    float* scores   = (float*)take((size_t)n * 4);
    int*   lut      = (int*)take((size_t)LUTSZ * 4);
    int*   slot     = (int*)take((size_t)n * 4);
    int*   cand     = (int*)take((size_t)n * 4);
    int*   sel      = (int*)take((size_t)n * 4);
    int*   hist     = (int*)take(4096 * 4);
    int*   counters = (int*)take(16);
    float* fsum     = (float*)take((size_t)KTOP * LATENT * 4);
    float* cnt      = (float*)take((size_t)KTOP * 4);
    (void)ws_size; (void)n_in; (void)out_size;

    const int TB = 256;
    int gb_n   = (n + TB - 1) / TB;
    int gb_lut = (LUTSZ + TB - 1) / TB;

    // re-init all state every launch (harness does not re-poison)
    fill_i32<<<gb_lut, TB, 0, stream>>>(lut, -1, LUTSZ);
    fill_i32<<<(4096 + TB - 1) / TB, TB, 0, stream>>>(hist, 0, 4096);
    fill_i32<<<1, 32, 0, stream>>>(counters, 0, 4);
    fill_f32<<<(KTOP * LATENT + TB - 1) / TB, TB, 0, stream>>>(fsum, 0.0f, KTOP * LATENT);
    fill_f32<<<1, TB, 0, stream>>>(cnt, 0.0f, KTOP);
    fill_f32<<<(4 * KTOP + TB - 1) / TB, TB, 0, stream>>>(out + o1, 0.0f, 4 * KTOP);
    fill_f32<<<(2 * KTOP + TB - 1) / TB, TB, 0, stream>>>(out + o3, 0.0f, 2 * KTOP);

    // pipeline
    gemv_wmma<<<(ntiles + 7) / 8, 256, 0, stream>>>(feats, conv_w, dotv, n, ntiles);
    head_kernel<<<gb_n, TB, 0, stream>>>(dotv, sem, conv_b, out, scores, n);
    scatter_lut<<<gb_n, TB, 0, stream>>>(coords, lut, n);
    peak_kernel<<<gb_n, TB, 0, stream>>>(coords, scores, lut, slot, cand, counters, n);
    hist_kernel<<<gb_n, TB, 0, stream>>>(cand, scores, counters, hist);
    thresh_kernel<<<1, 1, 0, stream>>>(hist, counters);
    compact_kernel<<<gb_n, TB, 0, stream>>>(cand, scores, counters, sel);
    rank_kernel<<<gb_n, TB, 0, stream>>>(sel, scores, coords, counters, slot, out, o1, o3, o4);
    owner_kernel<<<gb_n, TB, 0, stream>>>(coords, scores, lut, slot, feats, fsum, cnt, n);
    mean_kernel<<<(KTOP * LATENT + TB - 1) / TB, TB, 0, stream>>>(fsum, cnt, out + o2);
}
